// RN_29875792511277
// MI455X (gfx1250) — matrix operand order repaired and checked by reading, not compiled, not run on hardware
//
#include <hip/hip_runtime.h>
#include <hip/hip_bf16.h>
#include <math.h>

typedef __attribute__((ext_vector_type(16))) _Float16 v16h;
typedef __attribute__((ext_vector_type(8)))  _Float16 h8;
typedef __attribute__((ext_vector_type(8)))  float    v8f;

// ---------------------------------------------------------------------------
// Conv (3x3, stride2, pad1) + ReLU, direct form. <2% of total FLOPs.
// ---------------------------------------------------------------------------
__global__ void conv_relu_kernel(const float* __restrict__ in, const float* __restrict__ w,
                                 const float* __restrict__ bias, float* __restrict__ out,
                                 int B, int Cin, int Cout, int Hin, int Win, int Hout, int Wout)
{
    int idx = blockIdx.x * blockDim.x + threadIdx.x;
    int total = B * Cout * Hout * Wout;
    if (idx >= total) return;
    int x = idx % Wout; int t = idx / Wout;
    int y = t % Hout;  t /= Hout;
    int o = t % Cout;  int b = t / Cout;
    float acc = bias[o];
    for (int ci = 0; ci < Cin; ++ci) {
        const float* ip = in + ((b * Cin + ci) * Hin) * Win;
        const float* wp = w + ((o * Cin + ci) * 3) * 3;
        #pragma unroll
        for (int ky = 0; ky < 3; ++ky) {
            int iy = 2 * y - 1 + ky;
            if ((unsigned)iy >= (unsigned)Hin) continue;
            #pragma unroll
            for (int kx = 0; kx < 3; ++kx) {
                int ix = 2 * x - 1 + kx;
                if ((unsigned)ix >= (unsigned)Win) continue;
                acc += ip[iy * Win + ix] * wp[ky * 3 + kx];
            }
        }
    }
    out[idx] = fmaxf(acc, 0.f);
}

// Train-mode BatchNorm: per-channel batch stats (biased var), then apply.
__global__ void bn_stats_kernel(const float* __restrict__ x, float* __restrict__ stats,
                                int B, int C, int HW)
{
    int ch = blockIdx.x;
    int N = B * HW;
    float s = 0.f, s2 = 0.f;
    for (int i = threadIdx.x; i < N; i += blockDim.x) {
        int b = i / HW, p = i % HW;
        float v = x[(b * C + ch) * HW + p];
        s += v; s2 += v * v;
    }
    __shared__ float ls[256], ls2[256];
    ls[threadIdx.x] = s; ls2[threadIdx.x] = s2;
    __syncthreads();
    for (int off = 128; off > 0; off >>= 1) {
        if ((int)threadIdx.x < off) {
            ls[threadIdx.x]  += ls[threadIdx.x + off];
            ls2[threadIdx.x] += ls2[threadIdx.x + off];
        }
        __syncthreads();
    }
    if (threadIdx.x == 0) {
        float m = ls[0] / (float)N;
        float var = ls2[0] / (float)N - m * m;
        stats[ch * 2 + 0] = m;
        stats[ch * 2 + 1] = rsqrtf(var + 1e-5f);
    }
}

__global__ void bn_apply_kernel(float* __restrict__ x, const float* __restrict__ stats,
                                const float* __restrict__ g, const float* __restrict__ bb,
                                int B, int C, int HW)
{
    int idx = blockIdx.x * blockDim.x + threadIdx.x;
    int total = B * C * HW;
    if (idx >= total) return;
    int ch = (idx / HW) % C;
    float m = stats[ch * 2], r = stats[ch * 2 + 1];
    x[idx] = g[ch] * (x[idx] - m) * r + bb[ch];
}

// ---------------------------------------------------------------------------
// LSTM: one block per batch row (independent recurrences), 45 steps, h/c in LDS.
// Output = final CELL state c_n (matches the reference's (_, qst) destructure).
// ---------------------------------------------------------------------------
__global__ void lstm_kernel(const int* __restrict__ qidx, const float* __restrict__ emb,
                            const float* __restrict__ wih, const float* __restrict__ whh,
                            const float* __restrict__ bih, const float* __restrict__ bhh,
                            float* __restrict__ qout /* 64x128 */)
{
    int b = blockIdx.x;
    int j = threadIdx.x;                   // 0..511 (one gate column each)
    __shared__ float h[128], c[128], xt[32], gates[512];
    if (j < 128) { h[j] = 0.f; c[j] = 0.f; }
    __syncthreads();
    for (int t = 0; t < 45; ++t) {
        if (j < 32) {
            int w = qidx[b * 45 + t];
            xt[j] = emb[w * 32 + j];
        }
        __syncthreads();
        float g = bih[j] + bhh[j];
        const float* wi = wih + j * 32;
        #pragma unroll
        for (int k = 0; k < 32; ++k) g += xt[k] * wi[k];
        const float* wh = whh + j * 128;
        for (int k = 0; k < 128; ++k) g += h[k] * wh[k];
        gates[j] = g;
        __syncthreads();
        if (j < 128) {
            float ig = 1.f / (1.f + expf(-gates[j]));
            float fg = 1.f / (1.f + expf(-gates[128 + j]));
            float gg = tanhf(gates[256 + j]);
            float og = 1.f / (1.f + expf(-gates[384 + j]));
            float cn = fg * c[j] + ig * gg;
            c[j] = cn;
            h[j] = og * tanhf(cn);
        }
        __syncthreads();
    }
    if (j < 128) qout[b * 128 + j] = c[j];
}

// fp32 -> fp16 weight conversion
__global__ void f32_to_f16_kernel(const float* __restrict__ in, _Float16* __restrict__ out, int n)
{
    int i = blockIdx.x * blockDim.x + threadIdx.x;
    if (i < n) out[i] = (_Float16)in[i];
}

// ---------------------------------------------------------------------------
// g-layer-1 decomposition: U[b,p]=x_flat@W1[:, 0:26].T, V[b,p]=x_flat@W1[:,26:52].T
// (x_flat built on the fly from conv output + coords), Q[b]=qst@W1[:,52:180].T+gb1
// ---------------------------------------------------------------------------
__global__ void uv_kernel(const float* __restrict__ x4,   /* 64 x 24 x 64 */
                          const float* __restrict__ gw1,  /* 256 x 180 */
                          float* __restrict__ U, float* __restrict__ V)
{
    int idx = blockIdx.x * blockDim.x + threadIdx.x;      // 4096*256
    if (idx >= 4096 * 256) return;
    int j = idx & 255;
    int bp = idx >> 8;
    int b = bp >> 6, p = bp & 63;
    const float* wrow = gw1 + j * 180;
    float u = 0.f, v = 0.f;
    #pragma unroll
    for (int cc = 0; cc < 24; ++cc) {
        float xv = x4[(b * 24 + cc) * 64 + p];
        u += xv * wrow[cc];
        v += xv * wrow[26 + cc];
    }
    // cx = (p/8.0 - 4)/4 (float division!), cy = (p%8 - 4)/4
    float cx = (p * 0.125f - 4.0f) * 0.25f;
    float cy = ((float)(p & 7) - 4.0f) * 0.25f;
    u += cx * wrow[24] + cy * wrow[25];
    v += cx * wrow[50] + cy * wrow[51];
    U[idx] = u; V[idx] = v;
}

__global__ void qproj_kernel(const float* __restrict__ qst, const float* __restrict__ gw1,
                             const float* __restrict__ gb1, float* __restrict__ Q)
{
    int idx = blockIdx.x * blockDim.x + threadIdx.x;      // 64*256
    if (idx >= 64 * 256) return;
    int j = idx & 255, b = idx >> 8;
    const float* wrow = gw1 + j * 180 + 52;
    const float* qr = qst + b * 128;
    float s = gb1[j];
    for (int k = 0; k < 128; ++k) s += qr[k] * wrow[k];
    Q[idx] = s;
}

// h1[((b*64+a)*64+c), j] = relu(U[b*64+c,j] + V[b*64+a,j] + Q[b,j]) -> fp16
__global__ void h1_kernel(const float* __restrict__ U, const float* __restrict__ V,
                          const float* __restrict__ Q, _Float16* __restrict__ h1)
{
    int idx = blockIdx.x * blockDim.x + threadIdx.x;      // 262144*256 = 2^26
    int j = idx & 255;
    int r = idx >> 8;
    int cc = r & 63;
    int a = (r >> 6) & 63;
    int b = r >> 12;
    float v = U[(b * 64 + cc) * 256 + j] + V[(b * 64 + a) * 256 + j] + Q[b * 256 + j];
    h1[idx] = (_Float16)fmaxf(v, 0.f);
}

// ---------------------------------------------------------------------------
// WMMA g-layer: Out = relu(A[Mx256] @ W.T + bias), fp16 in/out, fp32 accum.
// Block = 256 threads = 8 waves; block covers 64 rows x 256 cols.
// Each wave owns a 32-column strip (2 B tiles) and register-blocks 4 row
// tiles against them (8 wmma per K-step, B fragments reused 4x), cutting
// weight re-fetch 4x vs a 16-row block. Waves are fully independent
// (no barriers; EXEC all-ones as WMMA requires).
// Fragment layouts per CDNA5 ISA 7.12.2 (16-bit A 16x32; B column-per-lane;
// C/D vgpr r / lane -> row m0+r+8*(lane>=16), col n0+(lane&15)).
// ---------------------------------------------------------------------------
__global__ void glayer_kernel(const _Float16* __restrict__ A,    // M x 256
                              const _Float16* __restrict__ W16,  // 256 x 256, [out][in]
                              const float* __restrict__ bias,    // 256
                              _Float16* __restrict__ Out, int M)
{
    int wave = threadIdx.x >> 5;
    int lane = threadIdx.x & 31;
    int mbase = blockIdx.x << 6;               // 64 rows per block
    if (mbase >= M) return;                    // uniform per block
    int n0 = wave << 5;                        // 32 columns per wave
    int lanehi = lane >> 4;                    // 0/1
    int lanelo = lane & 15;
    int khalf = lanehi * 8;
    const _Float16* Ap = A + (mbase + lanelo) * 256;                     // row tile 0 base
    const _Float16* W0 = W16 + (n0 + lanelo) * 256 + lanehi * 16;        // B tile 0
    const _Float16* W1 = W16 + (n0 + 16 + lanelo) * 256 + lanehi * 16;   // B tile 1
    v8f acc[4][2];
    #pragma unroll
    for (int rt = 0; rt < 4; ++rt)
        #pragma unroll
        for (int ct = 0; ct < 2; ++ct)
            acc[rt][ct] = (v8f){0.f, 0.f, 0.f, 0.f, 0.f, 0.f, 0.f, 0.f};

    #pragma unroll
    for (int k0 = 0; k0 < 256; k0 += 32) {
        // B fragments: each lane = one output column, 16 contiguous K values
        v16h b0 = *reinterpret_cast<const v16h*>(W0 + k0);
        v16h b1 = *reinterpret_cast<const v16h*>(W1 + k0);
        #pragma unroll
        for (int rt = 0; rt < 4; ++rt) {
            // A fragment (row tile rt): lanes<16 carry K=k0+[0..15];
            // lanes>=16 carry K=k0+16+[0..15].
            const _Float16* Art = Ap + rt * (16 * 256);
            h8 alo = *reinterpret_cast<const h8*>(Art + k0 + 2 * khalf);
            h8 ahi = *reinterpret_cast<const h8*>(Art + k0 + 2 * khalf + 8);
            v16h afrag;
            #pragma unroll
            for (int t = 0; t < 8; ++t) { afrag[t] = alo[t]; afrag[8 + t] = ahi[t]; }
            acc[rt][0] = __builtin_amdgcn_wmma_f32_16x16x32_f16(false, afrag, false, b0,
                                                                (short)0, acc[rt][0], false, false);
            acc[rt][1] = __builtin_amdgcn_wmma_f32_16x16x32_f16(false, afrag, false, b1,
                                                                (short)0, acc[rt][1], false, false);
        }
    }
    int col0 = n0 + lanelo;
    int col1 = n0 + 16 + lanelo;
    float b0v = bias[col0], b1v = bias[col1];
    #pragma unroll
    for (int rt = 0; rt < 4; ++rt) {
        int rbase = mbase + rt * 16 + 8 * lanehi;
        #pragma unroll
        for (int r = 0; r < 8; ++r) {
            int row = rbase + r;
            Out[row * 256 + col0] = (_Float16)fmaxf(acc[rt][0][r] + b0v, 0.f);
            Out[row * 256 + col1] = (_Float16)fmaxf(acc[rt][1][r] + b1v, 0.f);
        }
    }
}

// Sum-pool over 4096 pairs per batch: xg[b,j] = sum_r h4[b*4096+r, j]
__global__ void sumpool_kernel(const _Float16* __restrict__ h4, float* __restrict__ xg)
{
    int b = blockIdx.x; int j = threadIdx.x;
    const _Float16* p = h4 + b * 4096 * 256 + j;
    float s = 0.f;
    for (int r = 0; r < 4096; ++r) s += (float)p[r * 256];
    xg[b * 256 + j] = s;
}

// f-MLP (256->256->256->28) + log_softmax; one block per batch row.
__global__ void fmlp_kernel(const float* __restrict__ xg,
                            const float* __restrict__ fw1, const float* __restrict__ fb1,
                            const float* __restrict__ fw2, const float* __restrict__ fb2,
                            const float* __restrict__ fw3, const float* __restrict__ fb3,
                            float* __restrict__ out)
{
    int b = blockIdx.x; int j = threadIdx.x;
    __shared__ float s0[256], s1[256], lg[28];
    s0[j] = xg[b * 256 + j];
    __syncthreads();
    {
        const float* wr = fw1 + j * 256;
        float acc = fb1[j];
        for (int k = 0; k < 256; ++k) acc += s0[k] * wr[k];
        s1[j] = fmaxf(acc, 0.f);
    }
    __syncthreads();
    {
        const float* wr = fw2 + j * 256;
        float acc = fb2[j];
        for (int k = 0; k < 256; ++k) acc += s1[k] * wr[k];
        s0[j] = fmaxf(acc, 0.f);
    }
    __syncthreads();
    if (j < 28) {
        const float* wr = fw3 + j * 256;
        float acc = fb3[j];
        for (int k = 0; k < 256; ++k) acc += s0[k] * wr[k];
        lg[j] = acc;
    }
    __syncthreads();
    if (j == 0) {
        float mx = lg[0];
        for (int k = 1; k < 28; ++k) mx = fmaxf(mx, lg[k]);
        float se = 0.f;
        for (int k = 0; k < 28; ++k) se += expf(lg[k] - mx);
        float lse = mx + logf(se);
        for (int k = 0; k < 28; ++k) out[b * 28 + k] = lg[k] - lse;
    }
}

// ---------------------------------------------------------------------------
extern "C" void kernel_launch(void* const* d_in, const int* in_sizes, int n_in,
                              void* d_out, int out_size, void* d_ws, size_t ws_size,
                              hipStream_t stream)
{
    (void)in_sizes; (void)n_in; (void)out_size; (void)ws_size;
    const float* img  = (const float*)d_in[0];
    const int*   qidx = (const int*)d_in[1];
    const float* c1w = (const float*)d_in[2];  const float* c1b = (const float*)d_in[3];
    const float* c2w = (const float*)d_in[4];  const float* c2b = (const float*)d_in[5];
    const float* c3w = (const float*)d_in[6];  const float* c3b = (const float*)d_in[7];
    const float* c4w = (const float*)d_in[8];  const float* c4b = (const float*)d_in[9];
    const float* bn1g = (const float*)d_in[10]; const float* bn1b = (const float*)d_in[11];
    const float* bn2g = (const float*)d_in[12]; const float* bn2b = (const float*)d_in[13];
    const float* bn3g = (const float*)d_in[14]; const float* bn3b = (const float*)d_in[15];
    const float* bn4g = (const float*)d_in[16]; const float* bn4b = (const float*)d_in[17];
    const float* emb = (const float*)d_in[18];
    const float* wih = (const float*)d_in[19]; const float* whh = (const float*)d_in[20];
    const float* bih = (const float*)d_in[21]; const float* bhh = (const float*)d_in[22];
    const float* gw1 = (const float*)d_in[23]; const float* gb1 = (const float*)d_in[24];
    const float* gw2 = (const float*)d_in[25]; const float* gb2 = (const float*)d_in[26];
    const float* gw3 = (const float*)d_in[27]; const float* gb3 = (const float*)d_in[28];
    const float* gw4 = (const float*)d_in[29]; const float* gb4 = (const float*)d_in[30];
    const float* fw1 = (const float*)d_in[31]; const float* fb1 = (const float*)d_in[32];
    const float* fw2 = (const float*)d_in[33]; const float* fb2 = (const float*)d_in[34];
    const float* fw3 = (const float*)d_in[35]; const float* fb3 = (const float*)d_in[36];
    float* out = (float*)d_out;

    // ---- flat workspace layout (256B aligned slabs) ----
    char* ws = (char*)d_ws;
    size_t off = 0;
    auto alloc = [&](size_t bytes) -> char* {
        char* p = ws + off;
        off = (off + bytes + 255) & ~(size_t)255;
        return p;
    };
    float*    cb1   = (float*)alloc((size_t)64 * 24 * 64 * 64 * 4); // 25.2 MB
    float*    cb2   = (float*)alloc((size_t)64 * 24 * 32 * 32 * 4); //  6.3 MB
    float*    cb3   = (float*)alloc((size_t)64 * 24 * 16 * 16 * 4);
    float*    x4    = (float*)alloc((size_t)64 * 24 * 8 * 8 * 4);
    float*    stats = (float*)alloc(24 * 2 * 4);
    float*    qst   = (float*)alloc((size_t)64 * 128 * 4);
    float*    U     = (float*)alloc((size_t)4096 * 256 * 4);
    float*    V     = (float*)alloc((size_t)4096 * 256 * 4);
    float*    Q     = (float*)alloc((size_t)64 * 256 * 4);
    _Float16* w16a  = (_Float16*)alloc((size_t)256 * 256 * 2);
    _Float16* w16b  = (_Float16*)alloc((size_t)256 * 256 * 2);
    _Float16* w16c  = (_Float16*)alloc((size_t)256 * 256 * 2);
    _Float16* hA    = (_Float16*)alloc((size_t)262144 * 256 * 2);  // 134 MB (L2-resident)
    _Float16* hB    = (_Float16*)alloc((size_t)262144 * 256 * 2);  // 134 MB
    float*    xg    = (float*)alloc((size_t)64 * 256 * 4);

    // ---- conv stack: (conv->relu->bn) x4 ----
    {
        int n = 64 * 24 * 64 * 64;
        conv_relu_kernel<<<(n + 255) / 256, 256, 0, stream>>>(img, c1w, c1b, cb1, 64, 3, 24, 128, 128, 64, 64);
        bn_stats_kernel<<<24, 256, 0, stream>>>(cb1, stats, 64, 24, 64 * 64);
        bn_apply_kernel<<<(n + 255) / 256, 256, 0, stream>>>(cb1, stats, bn1g, bn1b, 64, 24, 64 * 64);
    }
    {
        int n = 64 * 24 * 32 * 32;
        conv_relu_kernel<<<(n + 255) / 256, 256, 0, stream>>>(cb1, c2w, c2b, cb2, 64, 24, 24, 64, 64, 32, 32);
        bn_stats_kernel<<<24, 256, 0, stream>>>(cb2, stats, 64, 24, 32 * 32);
        bn_apply_kernel<<<(n + 255) / 256, 256, 0, stream>>>(cb2, stats, bn2g, bn2b, 64, 24, 32 * 32);
    }
    {
        int n = 64 * 24 * 16 * 16;
        conv_relu_kernel<<<(n + 255) / 256, 256, 0, stream>>>(cb2, c3w, c3b, cb3, 64, 24, 24, 32, 32, 16, 16);
        bn_stats_kernel<<<24, 256, 0, stream>>>(cb3, stats, 64, 24, 16 * 16);
        bn_apply_kernel<<<(n + 255) / 256, 256, 0, stream>>>(cb3, stats, bn3g, bn3b, 64, 24, 16 * 16);
    }
    {
        int n = 64 * 24 * 8 * 8;
        conv_relu_kernel<<<(n + 255) / 256, 256, 0, stream>>>(cb3, c4w, c4b, x4, 64, 24, 24, 16, 16, 8, 8);
        bn_stats_kernel<<<24, 256, 0, stream>>>(x4, stats, 64, 24, 8 * 8);
        bn_apply_kernel<<<(n + 255) / 256, 256, 0, stream>>>(x4, stats, bn4g, bn4b, 64, 24, 8 * 8);
    }

    // ---- LSTM (qst = final cell state) ----
    lstm_kernel<<<64, 512, 0, stream>>>(qidx, emb, wih, whh, bih, bhh, qst);

    // ---- fp16 weight conversion for WMMA layers ----
    f32_to_f16_kernel<<<(65536 + 255) / 256, 256, 0, stream>>>(gw2, w16a, 65536);
    f32_to_f16_kernel<<<(65536 + 255) / 256, 256, 0, stream>>>(gw3, w16b, 65536);
    f32_to_f16_kernel<<<(65536 + 255) / 256, 256, 0, stream>>>(gw4, w16c, 65536);

    // ---- g layer 1 via decomposition, assemble h1 in fp16 ----
    uv_kernel<<<(4096 * 256) / 256, 256, 0, stream>>>(x4, gw1, U, V);
    qproj_kernel<<<(64 * 256) / 256, 256, 0, stream>>>(qst, gw1, gb1, Q);
    h1_kernel<<<262144, 256, 0, stream>>>(U, V, Q, hA);

    // ---- g layers 2-4: WMMA f16 GEMMs, 262144 x 256 x 256 each ----
    const int M = 262144;
    glayer_kernel<<<M / 64, 256, 0, stream>>>(hA, w16a, gb2, hB, M);
    glayer_kernel<<<M / 64, 256, 0, stream>>>(hB, w16b, gb3, hA, M);
    glayer_kernel<<<M / 64, 256, 0, stream>>>(hA, w16c, gb4, hB, M);

    // ---- sum-pool + f MLP + log_softmax ----
    sumpool_kernel<<<64, 256, 0, stream>>>(hB, xg);
    fmlp_kernel<<<64, 256, 0, stream>>>(xg, fw1, fb1, fw2, fb2, fw3, fb3, out);
}
